// LocalCausalAttention_89945205113447
// MI455X (gfx1250) — matrix operand order repaired and checked by reading, not compiled, not run on hardware
//
#include <hip/hip_runtime.h>
#include <hip/hip_bf16.h>

typedef __attribute__((ext_vector_type(16))) _Float16 v16h;
typedef __attribute__((ext_vector_type(8)))  float    v8f;
typedef __attribute__((ext_vector_type(2)))  _Float16 h2;

#define B_  2
#define N_  2048
#define D_  1024
#define H_  16
#define DH_ 64
#define WIN_ 64
#define SCALE_ 0.125f   // 1/sqrt(64)

// A-fragment K offset for 16-bit A 16x32 (ISA 7.12.2):
// VGPR j (0..3): K = half*8 + 2j ; VGPR j (4..7): K = 16 + half*8 + 2(j-4)
__device__ __forceinline__ int afk(int j, int half) {
    return ((j >> 2) << 4) + (half << 3) + ((j & 3) << 1);
}

__device__ __forceinline__ v8f wmma_f16(v16h a, v16h b, v8f c) {
    return __builtin_amdgcn_wmma_f32_16x16x32_f16(false, a, false, b, (short)0, c, false, false);
}

// ---------------- conversion kernels ----------------
__global__ void k_f32_to_f16(const float* __restrict__ in, _Float16* __restrict__ out, int n) {
    int i = blockIdx.x * blockDim.x + threadIdx.x;
    if (i < n) out[i] = (_Float16)in[i];
}

// LDS-tiled transpose+convert: in[R][C] f32 row-major -> out[C][R] f16.
// grid (C/32, R/32), block (32, 8). R, C multiples of 32.
__global__ __launch_bounds__(256) void k_transpose_f16(const float* __restrict__ in,
                                                       _Float16* __restrict__ out,
                                                       int R, int C) {
    __shared__ float tile[32][33];
    const int tx = threadIdx.x, ty = threadIdx.y;
    const int c0 = blockIdx.x * 32;
    const int r0 = blockIdx.y * 32;
#pragma unroll
    for (int i = ty; i < 32; i += 8)
        tile[i][tx] = in[(size_t)(r0 + i) * C + c0 + tx];
    __syncthreads();
#pragma unroll
    for (int i = ty; i < 32; i += 8)
        out[(size_t)(c0 + i) * R + r0 + tx] = (_Float16)tile[tx][i];
}

// Build ktr[b][h][d][n] (f16, n contiguous) from qkvh K columns.
// grid (DH/32, N/32, B*H), block (32, 8).
__global__ __launch_bounds__(256) void k_build_ktr(const _Float16* __restrict__ qkvh,
                                                   _Float16* __restrict__ ktr) {
    __shared__ _Float16 tile[32][33];
    const int tx = threadIdx.x, ty = threadIdx.y;
    const int bh = blockIdx.z;                 // b*16 + h
    const int d0 = blockIdx.x * 32;
    const int n0 = blockIdx.y * 32;
    const int b = bh >> 4, h = bh & 15;
    const _Float16* src = qkvh + (size_t)(b * N_) * (3 * D_) + D_ + h * DH_;
#pragma unroll
    for (int i = ty; i < 32; i += 8)
        tile[i][tx] = src[(size_t)(n0 + i) * (3 * D_) + d0 + tx];
    __syncthreads();
    _Float16* dst = ktr + (size_t)bh * DH_ * N_;
#pragma unroll
    for (int i = ty; i < 32; i += 8)
        dst[(size_t)(d0 + i) * N_ + n0 + tx] = tile[tx][i];
}

// ---------------- GEMM: C[M,N] = A[M,K] @ B[K,N] + bias[N] ----------------
// A row-major f16 (lda=K), B row-major f16 (ldb=N). 128 threads = 4 waves,
// each wave computes a 64x64 tile (4x4 WMMA subtiles), block tile = 128x128.
template <bool OUT_F16>
__global__ __launch_bounds__(128) void k_gemm_wmma(const _Float16* __restrict__ A,
                                                   const _Float16* __restrict__ Bm,
                                                   const float* __restrict__ bias,
                                                   void* __restrict__ Cout,
                                                   int M, int Nn, int K) {
    const int lane   = threadIdx.x & 31;
    const int wv     = threadIdx.x >> 5;
    const int lane15 = lane & 15;
    const int half   = lane >> 4;

    const int row0 = blockIdx.y * 128 + (wv >> 1) * 64;
    const int col0 = blockIdx.x * 128 + (wv & 1) * 64;

    v8f acc[4][4];
#pragma unroll
    for (int mt = 0; mt < 4; ++mt)
#pragma unroll
        for (int nt = 0; nt < 4; ++nt)
#pragma unroll
            for (int e = 0; e < 8; ++e) acc[mt][nt][e] = 0.0f;

    for (int k0 = 0; k0 < K; k0 += 32) {
        v16h af[4], bf[4];
#pragma unroll
        for (int mt = 0; mt < 4; ++mt) {
            const int m = row0 + mt * 16 + lane15;
            const _Float16* ap = A + (size_t)m * K + k0;
            __builtin_prefetch(ap + 32, 0, 3);     // next k-step -> global_prefetch_b8
#pragma unroll
            for (int j = 0; j < 8; ++j) {
                h2 t = *(const h2*)(ap + afk(j, half));
                af[mt][2 * j] = t[0];
                af[mt][2 * j + 1] = t[1];
            }
        }
        const _Float16* bp0 = Bm + (size_t)k0 * Nn + col0 + lane15;
#pragma unroll
        for (int nt = 0; nt < 4; ++nt) {
            const _Float16* bp = bp0 + nt * 16;
            __builtin_prefetch(bp + (size_t)32 * Nn, 0, 3);
#pragma unroll
            for (int j = 0; j < 8; ++j) {
                const int kr = afk(j, half);
                bf[nt][2 * j] = bp[(size_t)kr * Nn];
                bf[nt][2 * j + 1] = bp[(size_t)(kr + 1) * Nn];
            }
        }
#pragma unroll
        for (int mt = 0; mt < 4; ++mt)
#pragma unroll
            for (int nt = 0; nt < 4; ++nt)
                acc[mt][nt] = wmma_f16(af[mt], bf[nt], acc[mt][nt]);
    }

    // epilogue: C layout lane->N, VGPR j -> M = 8*half + j
#pragma unroll
    for (int nt = 0; nt < 4; ++nt) {
        const int n = col0 + nt * 16 + lane15;
        const float bv = bias[n];
#pragma unroll
        for (int mt = 0; mt < 4; ++mt) {
#pragma unroll
            for (int j = 0; j < 8; ++j) {
                const int m = row0 + mt * 16 + (half << 3) + j;
                const float v = acc[mt][nt][j] + bv;
                if (OUT_F16)
                    ((_Float16*)Cout)[(size_t)m * Nn + n] = (_Float16)v;
                else
                    ((float*)Cout)[(size_t)m * Nn + n] = v;
            }
        }
    }
}

// ---------------- windowed causal attention ----------------
// qkvh: [B*N][3072] f16 (q at col h*64, v at 2048+h*64)
// ktr : [B*H][DH][N] f16 (keys contiguous)
// ctxh: [B*N][1024] f16
// one wave per (b, h, 16-query tile); 8 waves per block
__global__ __launch_bounds__(256) void k_attn_wmma(const _Float16* __restrict__ qkvh,
                                                   const _Float16* __restrict__ ktr,
                                                   _Float16* __restrict__ ctxh) {
    const int lane   = threadIdx.x & 31;
    const int wv     = threadIdx.x >> 5;
    const int lane15 = lane & 15;
    const int half   = lane >> 4;

    const int gw = blockIdx.x * 8 + wv;            // 0..4095
    const int qt = gw & 127;                       // N/16
    const int h  = (gw >> 7) & 15;
    const int b  = gw >> 11;
    const int q0 = qt * 16;

    __shared__ _Float16 pbuf[8][16 * 96];
    _Float16* P = pbuf[wv];

    const size_t rs = 3 * 1024;                    // qkv row stride
    const _Float16* qbase = qkvh + (size_t)(b * N_) * rs + h * DH_;
    const _Float16* vbase = qbase + 2048;

    // Q A-fragments (16x32 per chunk, two chunks over DH=64)
    v16h aq[2];
#pragma unroll
    for (int c = 0; c < 2; ++c) {
        const _Float16* qp = qbase + (size_t)(q0 + lane15) * rs + c * 32;
#pragma unroll
        for (int j = 0; j < 8; ++j) {
            h2 t = *(const h2*)(qp + afk(j, half));
            aq[c][2 * j] = t[0];
            aq[c][2 * j + 1] = t[1];
        }
    }

    // K rows for this wave: ktr[(b*H+h)][d][:] ; contraction row d = c*32 + afk(j, half), key col = lane15
    const _Float16* krow0 = ktr + (size_t)((b * H_ + h) * DH_) * N_;               // c = 0
    const _Float16* krow1 = krow0 + (size_t)32 * N_;                               // c = 1

    // scores over 5 key tiles covering [q0-64, q0+16)
    v8f S[5];
#pragma unroll
    for (int t = 0; t < 5; ++t) {
        const int kb = q0 - 64 + t * 16;
        v16h bk[2];
        int kk = kb + lane15;
        kk = kk < 0 ? 0 : kk;                      // clamp; masked later
#pragma unroll
        for (int j = 0; j < 8; ++j) {
            const int d = afk(j, half);
            bk[0][2 * j] = krow0[(size_t)d * N_ + kk]; bk[0][2 * j + 1] = krow0[(size_t)(d + 1) * N_ + kk];
            bk[1][2 * j] = krow1[(size_t)d * N_ + kk]; bk[1][2 * j + 1] = krow1[(size_t)(d + 1) * N_ + kk];
        }
        v8f c;
#pragma unroll
        for (int e = 0; e < 8; ++e) c[e] = 0.0f;
        c = wmma_f16(aq[0], bk[0], c);
        c = wmma_f16(aq[1], bk[1], c);

        const int key = kb + lane15;               // C col = key
#pragma unroll
        for (int j = 0; j < 8; ++j) {
            const int q = q0 + (half << 3) + j;    // C row = query
            const int diff = q - key;
            const float s = c[j] * SCALE_;
            c[j] = (diff >= 0 && diff < WIN_ && key >= 0) ? s : -1.0e30f;
        }
        S[t] = c;
    }

    // softmax over keys (reduce across 16 lanes within each half)
    float mx[8], sm[8];
#pragma unroll
    for (int j = 0; j < 8; ++j) {
        float m = S[0][j];
#pragma unroll
        for (int t = 1; t < 5; ++t) m = fmaxf(m, S[t][j]);
        m = fmaxf(m, __shfl_xor(m, 1, 32));
        m = fmaxf(m, __shfl_xor(m, 2, 32));
        m = fmaxf(m, __shfl_xor(m, 4, 32));
        m = fmaxf(m, __shfl_xor(m, 8, 32));
        mx[j] = m;
        sm[j] = 0.0f;
    }
#pragma unroll
    for (int t = 0; t < 5; ++t)
#pragma unroll
        for (int j = 0; j < 8; ++j) {
            const float e = __expf(S[t][j] - mx[j]);
            S[t][j] = e;
            sm[j] += e;
        }
#pragma unroll
    for (int j = 0; j < 8; ++j) {
        float s = sm[j];
        s += __shfl_xor(s, 1, 32);
        s += __shfl_xor(s, 2, 32);
        s += __shfl_xor(s, 4, 32);
        s += __shfl_xor(s, 8, 32);
        sm[j] = 1.0f / s;
    }

    // stage P (16 queries x 96 keys, zero-padded 80..95) to LDS
#pragma unroll
    for (int t = 0; t < 5; ++t)
#pragma unroll
        for (int j = 0; j < 8; ++j)
            P[((half << 3) + j) * 96 + t * 16 + lane15] = (_Float16)(S[t][j] * sm[j]);
#pragma unroll
    for (int j = 0; j < 8; ++j)
        P[((half << 3) + j) * 96 + 80 + lane15] = (_Float16)0.0f;
    __syncthreads();

    // O = P[16x96] @ V[96x64]  -> 3 K-chunks x 4 N-tiles
    v8f o[4];
#pragma unroll
    for (int nt = 0; nt < 4; ++nt)
#pragma unroll
        for (int e = 0; e < 8; ++e) o[nt][e] = 0.0f;

#pragma unroll
    for (int kc = 0; kc < 3; ++kc) {
        v16h ap;
        const _Float16* pr = P + lane15 * 96 + kc * 32;
#pragma unroll
        for (int j = 0; j < 8; ++j) {
            h2 t = *(const h2*)(pr + afk(j, half));
            ap[2 * j] = t[0];
            ap[2 * j + 1] = t[1];
        }
        const int gb = q0 - 64 + kc * 32;          // key row = gb + afk(j, half)
#pragma unroll
        for (int nt = 0; nt < 4; ++nt) {
            v16h bv;
#pragma unroll
            for (int j = 0; j < 8; ++j) {
                int g0 = gb + afk(j, half), g1 = g0 + 1;
                g0 = g0 < 0 ? 0 : (g0 >= N_ ? N_ - 1 : g0);    // clamp; P==0 where invalid
                g1 = g1 < 0 ? 0 : (g1 >= N_ ? N_ - 1 : g1);
                bv[2 * j] = vbase[(size_t)g0 * rs + nt * 16 + lane15];
                bv[2 * j + 1] = vbase[(size_t)g1 * rs + nt * 16 + lane15];
            }
            o[nt] = wmma_f16(ap, bv, o[nt]);
        }
    }

    // store context: row (b*N + q), col h*64 + d
    _Float16* cp = ctxh + (size_t)(b * N_) * 1024 + h * DH_;
#pragma unroll
    for (int nt = 0; nt < 4; ++nt) {
        const int d = nt * 16 + lane15;
#pragma unroll
        for (int j = 0; j < 8; ++j) {
            const int q = q0 + (half << 3) + j;
            cp[(size_t)q * 1024 + d] = (_Float16)o[nt][j];
        }
    }
}

extern "C" void kernel_launch(void* const* d_in, const int* in_sizes, int n_in,
                              void* d_out, int out_size, void* d_ws, size_t ws_size,
                              hipStream_t stream) {
    const float* x    = (const float*)d_in[0];   // [B,N,D]
    const float* Wqkv = (const float*)d_in[1];   // [3D,D]
    const float* bqkv = (const float*)d_in[2];   // [3D]
    const float* Wout = (const float*)d_in[3];   // [D,D]
    const float* bout = (const float*)d_in[4];   // [D]
    float* out = (float*)d_out;

    const size_t MN = (size_t)B_ * N_;           // 4096
    _Float16* xh    = (_Float16*)d_ws;                       // 4096*1024
    _Float16* WqkvT = xh + MN * D_;                          // 1024*3072
    _Float16* WoutT = WqkvT + (size_t)D_ * 3 * D_;           // 1024*1024
    _Float16* qkvh  = WoutT + (size_t)D_ * D_;               // 4096*3072
    _Float16* ctxh  = qkvh + MN * 3 * D_;                    // 4096*1024
    _Float16* ktr   = ctxh + MN * D_;                        // B*H*DH*N = 4M

    // conversions
    {
        int n = (int)(MN * D_);
        k_f32_to_f16<<<(n + 255) / 256, 256, 0, stream>>>(x, xh, n);
    }
    k_transpose_f16<<<dim3(D_ / 32, 3 * D_ / 32), dim3(32, 8), 0, stream>>>(Wqkv, WqkvT, 3 * D_, D_);
    k_transpose_f16<<<dim3(D_ / 32, D_ / 32), dim3(32, 8), 0, stream>>>(Wout, WoutT, D_, D_);

    // QKV projection: [4096,1024] @ [1024,3072] -> f16 qkvh
    k_gemm_wmma<true><<<dim3(3 * D_ / 128, (int)(MN / 128)), 128, 0, stream>>>(
        xh, WqkvT, bqkv, (void*)qkvh, (int)MN, 3 * D_, D_);

    // K transpose for attention B-operand
    k_build_ktr<<<dim3(DH_ / 32, N_ / 32, B_ * H_), dim3(32, 8), 0, stream>>>(qkvh, ktr);

    // attention: 4096 waves, 8 waves/block
    k_attn_wmma<<<512, 256, 0, stream>>>(qkvh, ktr, ctxh);

    // output projection: [4096,1024] @ [1024,1024] + bout -> f32 d_out
    k_gemm_wmma<false><<<dim3(D_ / 128, (int)(MN / 128)), 128, 0, stream>>>(
        ctxh, WoutT, bout, (void*)out, (int)MN, D_, D_);
}
